// STransformer_73770358276526
// MI455X (gfx1250) — matrix operand order repaired and checked by reading, not compiled, not run on hardware
//
#include <hip/hip_runtime.h>

// ---------------- problem constants ----------------
constexpr int kB  = 4;
constexpr int kN  = 325;
constexpr int kT  = 36;
constexpr int kH  = 4;
constexpr int kBT = kB * kT;          // 144
constexpr int kRows = kBT * kN;       // 46800 token rows in [bt][n][c] layout
constexpr int kNP = 336;              // N padded to 21*16 for WMMA tiling
constexpr int kAdjLd = 328;           // adj_n row stride (multiple of 4 -> 16B rows)

typedef __bf16 bf16;
typedef __attribute__((ext_vector_type(16))) bf16 v16bf;
typedef __attribute__((ext_vector_type(8)))  float v8f;
typedef __attribute__((ext_vector_type(4)))  unsigned int v4u;

union FragU { v16bf v; v4u u[2]; };

#define WMMA_BF16(a, b, c) \
  __builtin_amdgcn_wmma_f32_16x16x32_bf16(false, (a), false, (b), (short)0, (c), false, false)

__device__ __forceinline__ bf16 tobf(float v) { return (bf16)v; }
__device__ __forceinline__ v8f zero8() {
  v8f z = {0.f, 0.f, 0.f, 0.f, 0.f, 0.f, 0.f, 0.f};
  return z;
}

// ---------------- LDS swizzle maps: tiles stored in fragment order ----------
// A tile (row r, k c) -> [mtile][lane][pos]; each lane's 16-bf16 operand is
// 32 contiguous bytes.
__device__ __forceinline__ int swzA(int r, int c) {
  int laneh = (c >> 3) & 1;
  int chunk = c >> 4;
  return ((r >> 4) * 32 + laneh * 16 + (r & 15)) * 16 + chunk * 8 + (c & 7);
}
// B tile 32x64 (k r, col c) -> [group][lane][pos]
__device__ __forceinline__ int swzB(int r, int c) {
  return (((c >> 4) * 32) + ((r >> 3) & 1) * 16 + (c & 15)) * 16 + (r >> 4) * 8 + (r & 7);
}

// load a full 16-bf16 fragment (32 contiguous bytes) from swizzled LDS
__device__ __forceinline__ v16bf ld_frag32(const bf16* buf, int tileIdx, int lane) {
  const bf16* p = buf + ((size_t)tileIdx * 32 + lane) * 16;
  FragU f;
  f.u[0] = *(const v4u*)p;
  f.u[1] = *(const v4u*)(p + 8);
  return f.v;
}

// A fragment with only K=0..15 live from [rows][16] row-major LDS
__device__ __forceinline__ v16bf ld_a16(const bf16* buf, int rowbase, int lane) {
  const bf16* p = buf + (size_t)(rowbase + (lane & 15)) * 16;
  FragU f;
  v4u z = {0u, 0u, 0u, 0u};
  f.u[0] = (lane < 16) ? *(const v4u*)p : z;
  f.u[1] = (lane < 16) ? *(const v4u*)(p + 8) : z;
  return f.v;
}
// B = X^T (scores): B[d][col] = xbuf[(colbase+col)*16 + d]; rows contiguous
__device__ __forceinline__ v16bf ld_bT16(const bf16* xbuf, int colbase, int lane) {
  const bf16* p = xbuf + (size_t)(colbase + (lane & 15)) * 16;
  v4u lo = *(const v4u*)p;
  v4u hi = *(const v4u*)(p + 8);
  v4u z = {0u, 0u, 0u, 0u};
  FragU f;
  f.u[0] = (lane < 16) ? lo : z;
  f.u[1] = (lane < 16) ? hi : z;
  return f.v;
}
// B = V from transposed LDS vsT[d][row]: K rows contiguous per lane
__device__ __forceinline__ v16bf ld_bV16(const bf16* vsT, int krowbase, int lane) {
  const bf16* p = vsT + (size_t)(lane & 15) * kNP + krowbase;
  v4u lo = *(const v4u*)p;
  v4u hi = *(const v4u*)(p + 8);
  v4u z = {0u, 0u, 0u, 0u};
  FragU f;
  f.u[0] = (lane < 16) ? lo : z;
  f.u[1] = (lane < 16) ? hi : z;
  return f.v;
}

// ---------------- generic batched WMMA GEMM ----------------
// C[M,Nc] = act(A[M,K] * B[K,Nc] + bias), row-major f32 in/out, bf16 compute.
// WG = 256 threads = 8 waves; 128x64 tile; wave -> 32x32 (2 M-tiles x 2 N-tiles,
// 4 WMMAs per k-stage).
__global__ __launch_bounds__(256) void k_gemm(const float* __restrict__ A,
                                              const float* __restrict__ Bm,
                                              float* __restrict__ Cout,
                                              int M, int Nc, int K,
                                              int lda, int ldb, int ldc,
                                              long long sA, long long sB, long long sC,
                                              const float* __restrict__ bias, int act) {
  __shared__ bf16 As[128 * 32];
  __shared__ bf16 Bs[32 * 64];
  A    += (long long)blockIdx.z * sA;
  Bm   += (long long)blockIdx.z * sB;
  Cout += (long long)blockIdx.z * sC;
  const int tid = threadIdx.x, wave = tid >> 5, lane = tid & 31;
  const int m0 = blockIdx.y * 128, n0 = blockIdx.x * 64;
  const int mtA = (wave & 3) * 2;
  const int mtB = mtA + 1;
  const int g0 = (wave >> 2) * 2;
  const bool aVec = ((lda & 3) == 0);
  const bool bVec = ((ldb & 3) == 0);
  v8f acc00 = zero8(), acc01 = zero8(), acc10 = zero8(), acc11 = zero8();
  for (int k0 = 0; k0 < K; k0 += 32) {
    const bool intA = aVec && (m0 + 128 <= M) && (k0 + 32 <= K);
    const bool intB = bVec && (k0 + 32 <= K) && (n0 + 64 <= Nc);
    if (intA) {                       // fast path: global b128 loads, no guards
#pragma unroll
      for (int i = 0; i < 4; ++i) {
        int e = tid + 256 * i;        // 1024 float4 chunks of the 128x32 tile
        int r = e >> 3;
        int c = (e & 7) * 4;
        const float4 v = *(const float4*)(A + (size_t)(m0 + r) * lda + k0 + c);
        bf16* d = &As[swzA(r, c)];    // 4 consecutive swizzled slots
        d[0] = tobf(v.x); d[1] = tobf(v.y); d[2] = tobf(v.z); d[3] = tobf(v.w);
      }
    } else {                          // boundary: clamped loads
#pragma unroll
      for (int i = 0; i < 16; ++i) {
        int e = tid + 256 * i;
        int r = e >> 5, c = e & 31;
        int gr = m0 + r, gc = k0 + c;
        int cr = (gr < M) ? gr : (M - 1);
        int cc = (gc < K) ? gc : (K - 1);
        float v = A[(size_t)cr * lda + cc];
        if (gr >= M || gc >= K) v = 0.f;
        As[swzA(r, c)] = tobf(v);
      }
    }
    if (intB) {
#pragma unroll
      for (int i = 0; i < 2; ++i) {
        int e = tid + 256 * i;        // 512 float4 chunks of the 32x64 tile
        int r = e >> 4;
        int c = (e & 15) * 4;
        const float4 v = *(const float4*)(Bm + (size_t)(k0 + r) * ldb + n0 + c);
        Bs[swzB(r, c)]     = tobf(v.x);
        Bs[swzB(r, c + 1)] = tobf(v.y);
        Bs[swzB(r, c + 2)] = tobf(v.z);
        Bs[swzB(r, c + 3)] = tobf(v.w);
      }
    } else {
#pragma unroll
      for (int i = 0; i < 8; ++i) {
        int e = tid + 256 * i;
        int r = e >> 6, c = e & 63;
        int gr = k0 + r, gc = n0 + c;
        int cr = (gr < K) ? gr : (K - 1);
        int cc = (gc < Nc) ? gc : (Nc - 1);
        float v = Bm[(size_t)cr * ldb + cc];
        if (gr >= K || gc >= Nc) v = 0.f;
        Bs[swzB(r, c)] = tobf(v);
      }
    }
    if (k0 + 32 < K) {                // prefetch next k-slab
      int pr = m0 + (tid & 127);
      if (pr < M) __builtin_prefetch(&A[(size_t)pr * lda + k0 + 32], 0, 0);
      int br = k0 + 32 + (tid & 31);
      if (br < K) __builtin_prefetch(&Bm[(size_t)br * ldb + n0], 0, 0);
    }
    __syncthreads();
    v16bf a0 = ld_frag32(As, mtA, lane);
    v16bf a1 = ld_frag32(As, mtB, lane);
    v16bf b0 = ld_frag32(Bs, g0, lane);
    v16bf b1 = ld_frag32(Bs, g0 + 1, lane);
    acc00 = WMMA_BF16(a0, b0, acc00);
    acc01 = WMMA_BF16(a0, b1, acc01);
    acc10 = WMMA_BF16(a1, b0, acc10);
    acc11 = WMMA_BF16(a1, b1, acc11);
    __syncthreads();
  }
  const int off = (lane < 16) ? 0 : 8;
  const int cl = lane & 15;
  const int gc0 = n0 + g0 * 16 + cl;
  const int gc1 = gc0 + 16;
  const float bv0 = (bias && gc0 < Nc) ? bias[gc0] : 0.f;
  const float bv1 = (bias && gc1 < Nc) ? bias[gc1] : 0.f;
#pragma unroll
  for (int r = 0; r < 8; ++r) {
    int gr0 = m0 + mtA * 16 + r + off;
    int gr1 = m0 + mtB * 16 + r + off;
    if (gr0 < M) {
      if (gc0 < Nc) {
        float v = acc00[r] + bv0;
        if (act) v = fmaxf(v, 0.f);
        Cout[(size_t)gr0 * ldc + gc0] = v;
      }
      if (gc1 < Nc) {
        float v = acc01[r] + bv1;
        if (act) v = fmaxf(v, 0.f);
        Cout[(size_t)gr0 * ldc + gc1] = v;
      }
    }
    if (gr1 < M) {
      if (gc0 < Nc) {
        float v = acc10[r] + bv0;
        if (act) v = fmaxf(v, 0.f);
        Cout[(size_t)gr1 * ldc + gc0] = v;
      }
      if (gc1 < Nc) {
        float v = acc11[r] + bv1;
        if (act) v = fmaxf(v, 0.f);
        Cout[(size_t)gr1 * ldc + gc1] = v;
      }
    }
  }
}

// ---------------- fused attention per (bt, head) ----------------
// S = (Q K^T)/8, column softmax (over q), O = P @ V. One WG per (bt,h);
// Q/K/V (325x16) live in LDS as bf16; V stored transposed for B-frag loads.
__global__ __launch_bounds__(256) void k_attn(const float* __restrict__ Qb,
                                              const float* __restrict__ Kb,
                                              const float* __restrict__ Vb,
                                              float* __restrict__ attn_out) {
  __shared__ bf16 qs[kNP * 16];
  __shared__ bf16 ks[kNP * 16];
  __shared__ bf16 vsT[16 * kNP];
  __shared__ bf16 ps[kNP * 16];
  __shared__ float redv[8 * 16];
  __shared__ float colmax[16];
  __shared__ float colsum[16];

  const int tid = threadIdx.x;
  const int wave = tid >> 5;
  const int lane = tid & 31;
  const int bh = blockIdx.x;                     // bt*H + h
  const size_t base = (size_t)bh * (kN * 16);

  // staging: float4 global loads, zero-pad rows >= N (branchless)
  for (int i = tid * 4; i < kNP * 16; i += 1024) {
    int row = i >> 4;
    int d = i & 15;
    int idx = (row < kN) ? i : 0;
    float4 q = *(const float4*)(Qb + base + idx);
    float4 k = *(const float4*)(Kb + base + idx);
    float4 v = *(const float4*)(Vb + base + idx);
    if (row >= kN) {
      q.x = q.y = q.z = q.w = 0.f;
      k.x = k.y = k.z = k.w = 0.f;
      v.x = v.y = v.z = v.w = 0.f;
    }
    bf16* qd = &qs[i];
    qd[0] = tobf(q.x); qd[1] = tobf(q.y); qd[2] = tobf(q.z); qd[3] = tobf(q.w);
    bf16* kd = &ks[i];
    kd[0] = tobf(k.x); kd[1] = tobf(k.y); kd[2] = tobf(k.z); kd[3] = tobf(k.w);
    vsT[(size_t)(d + 0) * kNP + row] = tobf(v.x);
    vsT[(size_t)(d + 1) * kNP + row] = tobf(v.y);
    vsT[(size_t)(d + 2) * kNP + row] = tobf(v.z);
    vsT[(size_t)(d + 3) * kNP + row] = tobf(v.w);
  }
  __syncthreads();

  int mts[3]; int nmt = 0;                       // this wave's q tiles (of 21)
  for (int i = 0; i < 3; ++i) { int t = wave + 8 * i; if (t < 21) mts[nmt++] = t; }
  const int off = (lane < 16) ? 0 : 8;
  const int col = lane & 15;

  v8f oacc[3];
  for (int t = 0; t < 3; ++t) oacc[t] = zero8();

  for (int kb = 0; kb < 21; ++kb) {              // 21 key-column blocks of 16
    v8f sfr[3];
    v16bf bkt = ld_bT16(ks, kb * 16, lane);      // B = K_block^T
    for (int t = 0; t < nmt; ++t) {
      v16bf a = ld_a16(qs, mts[t] * 16, lane);
      v8f z = zero8();
      z = WMMA_BF16(a, bkt, z);
#pragma unroll
      for (int r = 0; r < 8; ++r) z[r] *= 0.125f;   // 1/sqrt(C)
      sfr[t] = z;
    }
    // ---- column max over q (cross-wave) ----
    float m = -1e30f;
    for (int t = 0; t < nmt; ++t)
#pragma unroll
      for (int r = 0; r < 8; ++r) {
        int q = mts[t] * 16 + r + off;
        if (q < kN) m = fmaxf(m, sfr[t][r]);
      }
    m = fmaxf(m, __shfl_xor(m, 16));
    if (lane < 16) redv[wave * 16 + col] = m;
    __syncthreads();
    if (tid < 16) {
      float mm = -1e30f;
      for (int w = 0; w < 8; ++w) mm = fmaxf(mm, redv[w * 16 + tid]);
      colmax[tid] = mm;
    }
    __syncthreads();
    // ---- exp + column sum ----
    float cm = colmax[col];
    float psum = 0.f;
    for (int t = 0; t < nmt; ++t)
#pragma unroll
      for (int r = 0; r < 8; ++r) {
        int q = mts[t] * 16 + r + off;
        float e = (q < kN) ? __expf(sfr[t][r] - cm) : 0.f;
        sfr[t][r] = e;
        psum += e;
      }
    psum += __shfl_xor(psum, 16);
    if (lane < 16) redv[wave * 16 + col] = psum;
    __syncthreads();
    if (tid < 16) {
      float ss = 0.f;
      for (int w = 0; w < 8; ++w) ss += redv[w * 16 + tid];
      colsum[tid] = ss;
    }
    __syncthreads();
    // ---- normalize -> P (bf16 in LDS) ----
    float inv = 1.f / colsum[col];
    for (int t = 0; t < nmt; ++t)
#pragma unroll
      for (int r = 0; r < 8; ++r) {
        int q = mts[t] * 16 + r + off;
        ps[(size_t)q * 16 + col] = tobf(sfr[t][r] * inv);
      }
    __syncthreads();
    // ---- O += P_block @ V_block ----
    v16bf bv = ld_bV16(vsT, kb * 16, lane);
    for (int t = 0; t < nmt; ++t) {
      v16bf a = ld_a16(ps, mts[t] * 16, lane);
      oacc[t] = WMMA_BF16(a, bv, oacc[t]);
    }
    __syncthreads();
  }
  const int bt = bh >> 2;
  const int h = bh & 3;
  for (int t = 0; t < nmt; ++t)
#pragma unroll
    for (int r = 0; r < 8; ++r) {
      int q = mts[t] * 16 + r + off;
      if (q < kN)
        attn_out[((size_t)bt * kN + q) * 64 + h * 16 + col] = oacc[t][r];
    }
}

// ---------------- small VALU kernels ----------------
// instance-norm of adj into a row-padded (lda=328) buffer; pad cols zeroed
__global__ void k_adj_norm(const float* __restrict__ adj, float* __restrict__ adjn) {
  __shared__ float sred[64];
  __shared__ float stats[2];
  const int tid = threadIdx.x, wave = tid >> 5, lane = tid & 31;
  float s = 0.f, q = 0.f;
  for (int i = tid; i < kN * kN; i += 1024) { float v = adj[i]; s += v; q += v * v; }
  for (int m = 1; m < 32; m <<= 1) { s += __shfl_xor(s, m); q += __shfl_xor(q, m); }
  if (lane == 0) { sred[wave] = s; sred[32 + wave] = q; }
  __syncthreads();
  if (tid == 0) {
    float S = 0.f, Q = 0.f;
    for (int w = 0; w < 32; ++w) { S += sred[w]; Q += sred[32 + w]; }
    float mean = S / (float)(kN * kN);
    float var = Q / (float)(kN * kN) - mean * mean;
    stats[0] = mean; stats[1] = rsqrtf(var + 1e-5f);
  }
  __syncthreads();
  float mean = stats[0], rstd = stats[1];
  for (int i = tid; i < kN * kAdjLd; i += 1024) {
    int r = i / kAdjLd;
    int c = i - r * kAdjLd;
    float v = (c < kN) ? (adj[(size_t)r * kN + c] - mean) * rstd : 0.f;
    adjn[i] = v;
  }
}

__global__ void k_ds_embed(const float* __restrict__ adj, const float* __restrict__ eW,
                           const float* __restrict__ eb, float* __restrict__ DS) {
  int i = blockIdx.x * blockDim.x + threadIdx.x;
  if (i >= kN * 64) return;
  int c = i & 63;
  int n = i >> 6;
  float s = eb[c];
  const float* ar = adj + (size_t)n * kN;
  const float* wr = eW + (size_t)c * kN;
  for (int j = 0; j < kN; ++j) s += ar[j] * wr[j];
  DS[i] = s;
}

// [b][n][t][c] -> [bt][n][c]
__global__ void k_tr_in(const float* __restrict__ src, float* __restrict__ dst) {
  int i = blockIdx.x * blockDim.x + threadIdx.x;
  if (i >= kB * kN * kT * 64) return;
  int c = i & 63;
  int t = (i >> 6) % kT;
  int tmp = (i >> 6) / kT;
  int n = tmp % kN;
  int b = tmp / kN;
  dst[(((size_t)b * kT + t) * kN + n) * 64 + c] = src[i];
}

__global__ void k_tr_w(const float* __restrict__ src, float* __restrict__ dst, int R, int Cc) {
  int i = blockIdx.x * blockDim.x + threadIdx.x;
  if (i >= R * Cc) return;
  int r = i / Cc, c = i % Cc;
  dst[(size_t)c * R + r] = src[i];
}

// per-head Q/K/V projections (16x16 weights): Q,V from value+DS, K from key+DS
__global__ void k_qkv(const float* __restrict__ kTp, const float* __restrict__ vTp,
                      const float* __restrict__ DS,
                      const float* __restrict__ wq, const float* __restrict__ wk,
                      const float* __restrict__ wv,
                      float* __restrict__ Qb, float* __restrict__ Kb, float* __restrict__ Vb) {
  int i = blockIdx.x * blockDim.x + threadIdx.x;
  if (i >= kBT * kN * kH) return;
  int h = i & 3;
  int n = (i >> 2) % kN;
  int bt = (i >> 2) / kN;
  float vv[16], kk[16];
  size_t rb = ((size_t)bt * kN + n) * 64 + h * 16;
  size_t db = (size_t)n * 64 + h * 16;
#pragma unroll
  for (int e = 0; e < 16; ++e) {
    vv[e] = vTp[rb + e] + DS[db + e];
    kk[e] = kTp[rb + e] + DS[db + e];
  }
  size_t ob = (((size_t)bt * kH + h) * kN + n) * 16;
#pragma unroll
  for (int d = 0; d < 16; ++d) {
    float q = 0.f, k = 0.f, v = 0.f;
#pragma unroll
    for (int e = 0; e < 16; ++e) {
      q += vv[e] * wq[d * 16 + e];
      k += kk[e] * wk[d * 16 + e];
      v += vv[e] * wv[d * 16 + e];
    }
    Qb[ob + d] = q; Kb[ob + d] = k; Vb[ob + d] = v;
  }
}

// out = LN(x1 + x2 [+ DS(n)]) over C=64, one wave per row
__global__ void k_add_ln(const float* __restrict__ x1, const float* __restrict__ x2,
                         const float* __restrict__ ds,
                         const float* __restrict__ w, const float* __restrict__ b,
                         float* __restrict__ outp, int rows) {
  int row = blockIdx.x * 8 + (threadIdx.x >> 5);
  int lane = threadIdx.x & 31;
  if (row >= rows) return;
  int n = row % kN;
  size_t rb = (size_t)row * 64;
  float a0 = x1[rb + lane] + x2[rb + lane];
  float a1 = x1[rb + lane + 32] + x2[rb + lane + 32];
  if (ds) { a0 += ds[(size_t)n * 64 + lane]; a1 += ds[(size_t)n * 64 + lane + 32]; }
  float s = a0 + a1;
  for (int m = 1; m < 32; m <<= 1) s += __shfl_xor(s, m);
  float mean = s * (1.f / 64.f);
  float d0 = a0 - mean, d1 = a1 - mean;
  float q = d0 * d0 + d1 * d1;
  for (int m = 1; m < 32; m <<= 1) q += __shfl_xor(q, m);
  float rstd = rsqrtf(q * (1.f / 64.f) + 1e-5f);
  outp[rb + lane] = d0 * rstd * w[lane] + b[lane];
  outp[rb + lane + 32] = d1 * rstd * w[lane + 32] + b[lane + 32];
}

__global__ void k_logsoftmax(float* __restrict__ x, int rows) {
  int row = blockIdx.x * 8 + (threadIdx.x >> 5);
  int lane = threadIdx.x & 31;
  if (row >= rows) return;
  size_t rb = (size_t)row * 64;
  float a0 = x[rb + lane], a1 = x[rb + lane + 32];
  float m = fmaxf(a0, a1);
  for (int k = 1; k < 32; k <<= 1) m = fmaxf(m, __shfl_xor(m, k));
  float s = __expf(a0 - m) + __expf(a1 - m);
  for (int k = 1; k < 32; k <<= 1) s += __shfl_xor(s, k);
  float lse = m + __logf(s);
  x[rb + lane] = a0 - lse;
  x[rb + lane + 32] = a1 - lse;
}

// gated fusion + transpose back to [b][n][t][c]
__global__ void k_final(const float* __restrict__ us, const float* __restrict__ xg,
                        const float* __restrict__ gs, const float* __restrict__ gg,
                        float* __restrict__ outp) {
  int i = blockIdx.x * blockDim.x + threadIdx.x;
  if (i >= kBT * kN * 64) return;
  int c = i & 63;
  int n = (i >> 6) % kN;
  int bt = (i >> 6) / kN;
  int b = bt / kT, t = bt % kT;
  float g = 1.f / (1.f + __expf(-(gs[i] + gg[i])));
  float val = g * us[i] + (1.f - g) * xg[i];
  outp[(((size_t)b * kN + n) * kT + t) * 64 + c] = val;
}

// ---------------- host orchestration ----------------
extern "C" void kernel_launch(void* const* d_in, const int* in_sizes, int n_in,
                              void* d_out, int out_size, void* d_ws, size_t ws_size,
                              hipStream_t stream) {
  (void)in_sizes; (void)n_in; (void)out_size; (void)ws_size;
  const float* query   = (const float*)d_in[0];
  const float* key     = (const float*)d_in[1];
  const float* value   = (const float*)d_in[2];
  const float* adj     = (const float*)d_in[3];
  const float* embed_W = (const float*)d_in[4];
  const float* embed_b = (const float*)d_in[5];
  const float* wq      = (const float*)d_in[6];
  const float* wk      = (const float*)d_in[7];
  const float* wv      = (const float*)d_in[8];
  const float* fc_W    = (const float*)d_in[9];
  const float* fc_b    = (const float*)d_in[10];
  const float* n1_w    = (const float*)d_in[11];
  const float* n1_b    = (const float*)d_in[12];
  const float* n2_w    = (const float*)d_in[13];
  const float* n2_b    = (const float*)d_in[14];
  const float* ff1_W   = (const float*)d_in[15];
  const float* ff1_b   = (const float*)d_in[16];
  const float* ff2_W   = (const float*)d_in[17];
  const float* ff2_b   = (const float*)d_in[18];
  const float* gc1_W   = (const float*)d_in[19];
  const float* gc1_b   = (const float*)d_in[20];
  const float* gc2_W   = (const float*)d_in[21];
  const float* gc2_b   = (const float*)d_in[22];
  const float* fs_W    = (const float*)d_in[23];
  const float* fs_b    = (const float*)d_in[24];
  const float* fg_W    = (const float*)d_in[25];
  const float* fg_b    = (const float*)d_in[26];
  float* out = (float*)d_out;

  // ---- workspace layout (16B-aligned blocks, lifetime reuse) ----
  const size_t RNC = (size_t)kRows * 64;      // 2,995,200
  const size_t R2C = (size_t)kRows * 128;     // 5,990,400
  float* W = (float*)d_ws;
  size_t off = 0;
  auto alloc = [&](size_t n) {
    n = (n + 3) & ~(size_t)3;                 // keep every block 16B aligned
    float* p = W + off; off += n; return p;
  };
  float* adjn = alloc((size_t)kN * kAdjLd);   // padded rows, lda=328
  float* DS   = alloc((size_t)kN * 64);
  float* qT   = alloc(RNC);        // query, [bt][n][c]; residual later
  float* r3   = alloc(RNC);        // keyT -> attn_out -> gate_s
  float* r4   = alloc(RNC);        // valueT -> fc_out -> gate_g
  float* r5   = alloc(2 * R2C);    // s1(lo)+h(hi) -> ff1 activations (full)
  float* s2b  = alloc(RNC);
  float* xg   = alloc(RNC);
  float* r8   = alloc(RNC);        // Qb -> M_s
  float* r9   = alloc(RNC);        // Kb -> ff2 out
  float* r10  = alloc(RNC);        // Vb -> U_s
  float* fcWt  = alloc(64 * 64);
  float* ff1Wt = alloc(64 * 256);
  float* ff2Wt = alloc(256 * 64);
  float* fsWt  = alloc(64 * 64);
  float* fgWt  = alloc(64 * 64);
  float* s1 = r5;
  float* hB = r5 + R2C;

  auto gemm = [&](const float* A, const float* Bm, float* Cc, int M, int Nc, int K,
                  int lda, int ldb, int ldc, int batch, long long sA, long long sB,
                  long long sC, const float* bias, int act) {
    dim3 g((Nc + 63) / 64, (M + 127) / 128, batch);
    k_gemm<<<g, dim3(256), 0, stream>>>(A, Bm, Cc, M, Nc, K, lda, ldb, ldc,
                                        sA, sB, sC, bias, act);
  };

  const int eltBlocks = (int)((RNC + 255) / 256);

  // 1) adj instance-norm + spatial embedding
  k_adj_norm<<<1, 1024, 0, stream>>>(adj, adjn);
  k_ds_embed<<<(kN * 64 + 255) / 256, 256, 0, stream>>>(adj, embed_W, embed_b, DS);

  // 2) layout transposes  [b,n,t,c] -> [bt,n,c]
  k_tr_in<<<eltBlocks, 256, 0, stream>>>(query, qT);
  k_tr_in<<<eltBlocks, 256, 0, stream>>>(key, r3);
  k_tr_in<<<eltBlocks, 256, 0, stream>>>(value, r4);

  // 3) weight transposes for x @ W^T GEMMs
  k_tr_w<<<(64 * 64 + 255) / 256, 256, 0, stream>>>(fc_W, fcWt, 64, 64);
  k_tr_w<<<(256 * 64 + 255) / 256, 256, 0, stream>>>(ff1_W, ff1Wt, 256, 64);
  k_tr_w<<<(64 * 256 + 255) / 256, 256, 0, stream>>>(ff2_W, ff2Wt, 64, 256);
  k_tr_w<<<(64 * 64 + 255) / 256, 256, 0, stream>>>(fs_W, fsWt, 64, 64);
  k_tr_w<<<(64 * 64 + 255) / 256, 256, 0, stream>>>(fg_W, fgWt, 64, 64);

  // 4) GCN branch  (adjn GEMMs now take the vectorized A path: lda=328)
  gemm(qT, gc1_W, s1, kRows, 128, 64, 64, 128, 128, 1, 0, 0, 0, nullptr, 0);
  gemm(adjn, s1, hB, kN, 128, kN, kAdjLd, 128, 128, kBT, 0, (long long)kN * 128,
       (long long)kN * 128, gc1_b, 1);
  gemm(hB, gc2_W, s2b, kRows, 64, 128, 128, 64, 64, 1, 0, 0, 0, nullptr, 0);
  gemm(adjn, s2b, xg, kN, 64, kN, kAdjLd, 64, 64, kBT, 0, (long long)kN * 64,
       (long long)kN * 64, gc2_b, 0);
  k_logsoftmax<<<kRows / 8, 256, 0, stream>>>(xg, kRows);

  // 5) attention branch
  k_qkv<<<(kBT * kN * kH + 255) / 256, 256, 0, stream>>>(r3, r4, DS, wq, wk, wv,
                                                         r8, r9, r10);
  k_attn<<<kBT * kH, 256, 0, stream>>>(r8, r9, r10, r3);  // r3 = attention out
  gemm(r3, fcWt, r4, kRows, 64, 64, 64, 64, 64, 1, 0, 0, 0, fc_b, 0);
  k_add_ln<<<kRows / 8, 256, 0, stream>>>(r4, qT, DS, n1_w, n1_b, r8, kRows); // M_s

  // 6) FFN + LN2
  gemm(r8, ff1Wt, r5, kRows, 256, 64, 64, 256, 256, 1, 0, 0, 0, ff1_b, 1);
  gemm(r5, ff2Wt, r9, kRows, 64, 256, 256, 64, 64, 1, 0, 0, 0, ff2_b, 0);
  k_add_ln<<<kRows / 8, 256, 0, stream>>>(r9, r8, nullptr, n2_w, n2_b, r10, kRows); // U_s

  // 7) gated fusion
  gemm(r10, fsWt, r3, kRows, 64, 64, 64, 64, 64, 1, 0, 0, 0, fs_b, 0);
  gemm(xg, fgWt, r4, kRows, 64, 64, 64, 64, 64, 1, 0, 0, 0, fg_b, 0);
  k_final<<<eltBlocks, 256, 0, stream>>>(r10, xg, r3, r4, out);
}
